// TLModel_35862976922337
// MI455X (gfx1250) — matrix operand order repaired and checked by reading, not compiled, not run on hardware
//
#include <hip/hip_runtime.h>
#include <hip/hip_bf16.h>
#include <math.h>

#define NTOKEN 50000
#define NINP   400
#define NHID   1150
#define SEQ    70
#define BSZ    32
#define NCAND  33
#define TTOT   (SEQ*BSZ)          /* 2240 */
#define KPAD   416                /* NINP padded: 13 k-tiles of 32 */
#define NPAD   1152               /* NHID padded: 72 n-tiles of 16, 36 k-tiles of 32 */
#define KT_X   (KPAD/32)          /* 13 */
#define KT_H   (NPAD/32)          /* 36 */
#define NT     (NPAD/16)          /* 72 */
#define MT_C   ((TTOT*NCAND)/16)  /* 4620 candidate M-tiles */

#define BETA_T   0.5f
#define THR_R    10.0f
#define THR_TEMP 10.0f
#define BIGINF   1e8f
#define EPS_D    1e-8f

typedef __attribute__((ext_vector_type(16))) __bf16 v16bf;
typedef __attribute__((ext_vector_type(8)))  float  v8f;

#if defined(__has_builtin)
#if __has_builtin(__builtin_amdgcn_tensor_load_to_lds) && \
    __has_builtin(__builtin_amdgcn_s_wait_tensorcnt)
#define HAVE_TDM 1
#endif
#endif

static __device__ __forceinline__ v8f wmma_bf16(v16bf a, v16bf b, v8f c) {
  // D = A(16x32 bf16) * B(32x16 bf16) + C(16x16 f32)
  return __builtin_amdgcn_wmma_f32_16x16x32_bf16(false, a, false, b, (short)0, c,
                                                 false, false);
}

static __device__ __forceinline__ v8f zero8() {
  v8f z;
#pragma unroll
  for (int i = 0; i < 8; ++i) z[i] = 0.f;
  return z;
}

// A fragment (16x32 tile, row-major source, element stride ldk), bf16 source.
// lane l: row = l&15; kb = (l<16)?0:8; frag[e] = A[row, kb + e] (e<8)
//                                      frag[8+e] = A[row, kb + 16 + e]
static __device__ __forceinline__ v16bf a_frag_bf16(const __bf16* base, int ldk, int lane) {
  const int row = lane & 15;
  const int kb  = (lane < 16) ? 0 : 8;
  const __bf16* p = base + (size_t)row * ldk + kb;
  v16bf f;
#pragma unroll
  for (int e = 0; e < 8; ++e) f[e] = p[e];
#pragma unroll
  for (int e = 0; e < 8; ++e) f[8 + e] = p[16 + e];
  return f;
}

// Same A fragment but converting from f32 source on the fly.
static __device__ __forceinline__ v16bf a_frag_f32(const float* base, int ldk, int lane) {
  const int row = lane & 15;
  const int kb  = (lane < 16) ? 0 : 8;
  const float* p = base + (size_t)row * ldk + kb;
  v16bf f;
#pragma unroll
  for (int e = 0; e < 8; ++e) f[e] = (__bf16)p[e];
#pragma unroll
  for (int e = 0; e < 8; ++e) f[8 + e] = (__bf16)p[16 + e];
  return f;
}

// B fragment (32x16 tile). Source packed n-major: W[n][k], stride ldk.
// lane l: col = l&15; kb = (l<16)?0:8; frag[e] = W[col, kb + e] (e<8), frag[8+e] = W[col, kb + 16 + e]
static __device__ __forceinline__ v16bf b_frag_bf16(const __bf16* base, int ldk, int lane) {
  const int col = lane & 15;
  const int kb  = (lane < 16) ? 0 : 8;
  const __bf16* p = base + (size_t)col * ldk + kb;
  v16bf f;
#pragma unroll
  for (int e = 0; e < 8; ++e) f[e] = p[e];
#pragma unroll
  for (int e = 0; e < 8; ++e) f[8 + e] = p[16 + e];
  return f;
}

static __device__ __forceinline__ float eluf(float x) {
  return x > 0.f ? x : (__expf(x) - 1.f);
}

#ifdef HAVE_TDM
typedef __attribute__((ext_vector_type(4))) unsigned int u32x4;
typedef __attribute__((ext_vector_type(8))) int          i32x8;
typedef __attribute__((ext_vector_type(4))) int          i32x4;

// Issue a TDM DMA of one contiguous B tile (16*KPAD bf16 = 13312 B) into LDS.
// Descriptor per CDNA5 ISA ch.8: 1-D tile, data_size=2B, count=1, type=2 (image).
// This toolchain exposes the 6-arg builtin: (g0, g1, g2, g3, g4, cpol).
static __device__ __forceinline__ void tdm_load_tile(const __bf16* gsrc, __bf16* ldst) {
  const unsigned elems = 16u * KPAD;  // 6656 elements, contiguous
  unsigned long long ga = (unsigned long long)(size_t)gsrc;
  unsigned lds = (unsigned)(size_t)ldst;  // low 32 bits of shared pointer = LDS byte addr
  u32x4 g0;
  g0[0] = 1u;                                            // count=1 (valid), user mode
  g0[1] = lds;                                           // lds_addr
  g0[2] = (unsigned)ga;                                  // global_addr[31:0]
  g0[3] = (unsigned)((ga >> 32) & 0x01FFFFFFull) | (2u << 30);  // addr[56:32] | type=2
  i32x8 g1;
  g1[0] = (int)(1u << 16);                               // data_size=1 -> 2 bytes
  g1[1] = (int)((elems & 0xFFFFu) << 16);                // tensor_dim0[15:0]
  g1[2] = (int)((elems >> 16) | (1u << 16));             // tensor_dim0[31:16] | tensor_dim1=1
  g1[3] = (int)((elems & 0xFFFFu) << 16);                // tile_dim0 = elems
  g1[4] = 1;                                             // tile_dim1 = 1
  g1[5] = (int)elems;                                    // tensor_dim0_stride[31:0]
  g1[6] = 0;
  g1[7] = 0;
  i32x4 z4;
  z4[0] = 0; z4[1] = 0; z4[2] = 0; z4[3] = 0;
  i32x8 z8;
#pragma unroll
  for (int i = 0; i < 8; ++i) z8[i] = 0;
  __builtin_amdgcn_tensor_load_to_lds(g0, g1, z4, z4, z8, 0);
}
#endif

// ---------------------------------------------------------------- pack kernels

// W [rows, cols] f32  ->  Wp [prows, pcols] bf16, zero padded.
__global__ void k_pack_w(const float* __restrict__ W, __bf16* __restrict__ Wp,
                         int rows, int cols, int prows, int pcols) {
  int idx = blockIdx.x * blockDim.x + threadIdx.x;
  int total = prows * pcols;
  if (idx >= total) return;
  int r = idx / pcols, c = idx - r * pcols;
  float v = (r < rows && c < cols) ? W[(size_t)r * cols + c] : 0.f;
  Wp[idx] = (__bf16)v;
}

// emb_bf16[r][k] = encoder[data[r]][k] (k < NINP), zero padded to KPAD.
__global__ void k_embed_pack(const int* __restrict__ data, const float* __restrict__ enc,
                             __bf16* __restrict__ emb) {
  int idx = blockIdx.x * blockDim.x + threadIdx.x;
  int total = TTOT * KPAD;
  if (idx >= total) return;
  int r = idx / KPAD, k = idx - r * KPAD;
  int tok = data[r];
  float v = (k < NINP) ? enc[(size_t)tok * NINP + k] : 0.f;
  emb[idx] = (__bf16)v;
}

// h_hist rows 0..31 = h0, padded to NPAD cols.
__global__ void k_h0_init(const float* __restrict__ h0, float* __restrict__ h_hist) {
  int idx = blockIdx.x * blockDim.x + threadIdx.x;
  int total = BSZ * NPAD;
  if (idx >= total) return;
  int r = idx / NPAD, n = idx - r * NPAD;
  h_hist[idx] = (n < NHID) ? h0[(size_t)r * NHID + n] : 0.f;
}

// ------------------------------------------------------------- RNN step (x70)
// h_next = elu(emb_s @ W_ih^T + b_ih + h_prev @ W_hh^T + b_hh)
// grid: 72 n-tiles, block: 64 threads (2 waves -> batch rows 0-15 / 16-31)
__global__ void k_rnn_step(const __bf16* __restrict__ emb,
                           const __bf16* __restrict__ Wih,
                           const __bf16* __restrict__ Whh,
                           const float* __restrict__ b_ih,
                           const float* __restrict__ b_hh,
                           float* __restrict__ h_hist, int s) {
  const int ntile = blockIdx.x;
  const int wave  = threadIdx.x >> 5;
  const int lane  = threadIdx.x & 31;

  const __bf16* embA  = emb + (size_t)(s * BSZ + wave * 16) * KPAD;
  const float*  hprev = h_hist + (size_t)s * BSZ * NPAD + (size_t)(wave * 16) * NPAD;
  const __bf16* Bih   = Wih + (size_t)ntile * 16 * KPAD;
  const __bf16* Bhh   = Whh + (size_t)ntile * 16 * NPAD;

  v8f acc = zero8();
  for (int kt = 0; kt < KT_X; ++kt)
    acc = wmma_bf16(a_frag_bf16(embA + kt * 32, KPAD, lane),
                    b_frag_bf16(Bih + kt * 32, KPAD, lane), acc);
  for (int kt = 0; kt < KT_H; ++kt)
    acc = wmma_bf16(a_frag_f32(hprev + kt * 32, NPAD, lane),
                    b_frag_bf16(Bhh + kt * 32, NPAD, lane), acc);

  const int n    = ntile * 16 + (lane & 15);
  const float bias = (n < NHID) ? (b_ih[n] + b_hh[n]) : 0.f;
  const int moff = wave * 16 + ((lane < 16) ? 0 : 8);
  float* hnext = h_hist + (size_t)(s + 1) * BSZ * NPAD;
#pragma unroll
  for (int v = 0; v < 8; ++v) {
    float val = (n < NHID) ? eluf(acc[v] + bias) : 0.f;
    hnext[(size_t)(moff + v) * NPAD + n] = val;
  }
}

// ----------------------------------------------------- hU = h_flat @ W_hh^T + b_hh
// grid: (140 m-tiles, 18), block 128 (4 waves -> 72 n-tiles)
__global__ void k_hu_gemm(const float* __restrict__ h_flat,
                          const __bf16* __restrict__ Whh,
                          const float* __restrict__ b_hh,
                          float* __restrict__ hU) {
  const int mtile = blockIdx.x;
  const int wave  = threadIdx.x >> 5;
  const int lane  = threadIdx.x & 31;
  const int ntile = blockIdx.y * 4 + wave;

  const float*  A = h_flat + (size_t)mtile * 16 * NPAD;
  const __bf16* B = Whh + (size_t)ntile * 16 * NPAD;

  v8f acc = zero8();
  for (int kt = 0; kt < KT_H; ++kt)
    acc = wmma_bf16(a_frag_f32(A + kt * 32, NPAD, lane),
                    b_frag_bf16(B + kt * 32, NPAD, lane), acc);

  const int n = ntile * 16 + (lane & 15);
  const float bias = (n < NHID) ? b_hh[n] : 0.f;
  const int off = (lane < 16) ? 0 : 8;
#pragma unroll
  for (int v = 0; v < 8; ++v)
    hU[(size_t)(mtile * 16 + off + v) * NPAD + n] = acc[v] + bias;
}

// --------------------------------------- fused candidate GEMM + distance + score
// Block = 64 threads (2 waves, 2 M-tiles). B tiles (16 x KPAD bf16, contiguous)
// are double-buffered in LDS: wave 0 drives the Tensor Data Mover to DMA tile
// nt+1 while both waves run WMMAs out of tile nt (s_wait_tensorcnt + barrier).
// A tiles (gathered candidate embeddings, f32->bf16) staged per wave in LDS.
// grid: 2310 blocks -> 4620 M-tiles exactly.
__global__ void k_cand_score(const int* __restrict__ cand,
                             const float* __restrict__ enc,
                             const __bf16* __restrict__ Wih,
                             const float* __restrict__ b_ih,
                             const float* __restrict__ hU,
                             const float* __restrict__ h_flat,
                             const float* __restrict__ b_w,
                             const float* __restrict__ b_h,
                             const float* __restrict__ dec_bias,
                             float* __restrict__ scores) {
  __shared__ __align__(16) __bf16 lds_a[2][16 * KPAD];   // 26 KB
  __shared__ __align__(16) __bf16 lds_b[2][16 * KPAD];   // 26 KB (double buffer)

  const int wave  = threadIdx.x >> 5;
  const int lane  = threadIdx.x & 31;
  const int mtile = blockIdx.x * 2 + wave;
  const int rbase = mtile * 16;

  // Stage A: gather 16 candidate embeddings, convert f32 -> bf16, pad K.
  __bf16* la = lds_a[wave];
  for (int idx = lane; idx < 16 * KPAD; idx += 32) {
    int row = idx / KPAD, k = idx - row * KPAD;
    int tok = cand[rbase + row];
    float v = (k < NINP) ? enc[(size_t)tok * NINP + k] : 0.f;
    la[idx] = (__bf16)v;
  }

  const int off = (lane < 16) ? 0 : 8;
  int tv[8];
#pragma unroll
  for (int v = 0; v < 8; ++v) tv[v] = (rbase + off + v) / NCAND;

  float pd[8];
#pragma unroll
  for (int v = 0; v < 8; ++v) pd[v] = 0.f;

  // Prologue: bring in B tile 0.
#ifdef HAVE_TDM
  if (wave == 0) tdm_load_tile(Wih, lds_b[0]);
#else
  {
    const uint4* src = (const uint4*)Wih;
    uint4* dst = (uint4*)lds_b[0];
    for (int i = threadIdx.x; i < (16 * KPAD) / 8; i += 64) dst[i] = src[i];
  }
#endif

  for (int nt = 0; nt < NT; ++nt) {
#ifdef HAVE_TDM
    if (wave == 0) __builtin_amdgcn_s_wait_tensorcnt(0);
#endif
    __syncthreads();  // B tile nt resident; all waves done reading buffer nt+1's slot

#ifdef HAVE_TDM
    if (wave == 0 && nt + 1 < NT)
      tdm_load_tile(Wih + (size_t)(nt + 1) * 16 * KPAD, lds_b[(nt + 1) & 1]);
#else
    if (nt + 1 < NT) {
      const uint4* src = (const uint4*)(Wih + (size_t)(nt + 1) * 16 * KPAD);
      uint4* dst = (uint4*)lds_b[(nt + 1) & 1];
      for (int i = threadIdx.x; i < (16 * KPAD) / 8; i += 64) dst[i] = src[i];
    }
#endif

    const __bf16* lb = lds_b[nt & 1];
    v8f acc = zero8();
    for (int kt = 0; kt < KT_X; ++kt)
      acc = wmma_bf16(a_frag_bf16(la + kt * 32, KPAD, lane),
                      b_frag_bf16(lb + kt * 32, KPAD, lane), acc);

    const int n = nt * 16 + (lane & 15);
    const float bih = (n < NHID) ? b_ih[n] : 0.f;
#pragma unroll
    for (int v = 0; v < 8; ++v) {
      float hu = hU[(size_t)tv[v] * NPAD + n];
      float hf = h_flat[(size_t)tv[v] * NPAD + n];
      float e  = eluf(acc[v] + bih + hu);
      float df = hf - e;
      pd[v] += df * df;
    }
  }

  // Reduce over the 16 lanes sharing each matrix row, then apply the score chain.
  const float bh0 = b_h[0];
#pragma unroll
  for (int v = 0; v < 8; ++v) {
    float s = pd[v];
    s += __shfl_xor(s, 1, 32);
    s += __shfl_xor(s, 2, 32);
    s += __shfl_xor(s, 4, 32);
    s += __shfl_xor(s, 8, 32);
    if ((lane & 15) == 0) {
      int r = rbase + off + v;
      int tok = cand[r];
      float d = sqrtf(s + EPS_D);
      d += b_w[tok] + bh0;
      d += BIGINF / (1.f + __expf(-THR_TEMP * (d - THR_R)));
      d *= BETA_T;
      d += dec_bias[tok];
      scores[r] = -d;
    }
  }
}

// ------------------------------------------------- log-softmax over 33 candidates
// grid: 560 blocks x 128 threads; one warp per row t.
__global__ void k_logsoftmax(const float* __restrict__ sc, float* __restrict__ out) {
  const int wave = threadIdx.x >> 5;
  const int lane = threadIdx.x & 31;
  const int t = blockIdx.x * 4 + wave;
  const float* row = sc + (size_t)t * NCAND;

  float x  = row[lane];                                  // c = lane (0..31)
  float x2 = (lane == 0) ? row[32] : -INFINITY;          // c = 32 on lane 0

  float mx = fmaxf(x, x2);
  mx = fmaxf(mx, __shfl_xor(mx, 1, 32));
  mx = fmaxf(mx, __shfl_xor(mx, 2, 32));
  mx = fmaxf(mx, __shfl_xor(mx, 4, 32));
  mx = fmaxf(mx, __shfl_xor(mx, 8, 32));
  mx = fmaxf(mx, __shfl_xor(mx, 16, 32));

  float s = __expf(x - mx) + ((lane == 0) ? __expf(x2 - mx) : 0.f);
  s += __shfl_xor(s, 1, 32);
  s += __shfl_xor(s, 2, 32);
  s += __shfl_xor(s, 4, 32);
  s += __shfl_xor(s, 8, 32);
  s += __shfl_xor(s, 16, 32);

  float lse = mx + __logf(s);
  out[(size_t)t * NCAND + lane] = x - lse;
  if (lane == 0) out[(size_t)t * NCAND + 32] = x2 - lse;
}

// ---------------------------------------------------------------------- launch

extern "C" void kernel_launch(void* const* d_in, const int* in_sizes, int n_in,
                              void* d_out, int out_size, void* d_ws, size_t ws_size,
                              hipStream_t stream) {
  const int*   data     = (const int*)d_in[0];
  const int*   cand     = (const int*)d_in[1];
  const float* enc      = (const float*)d_in[2];
  const float* W_ih     = (const float*)d_in[3];
  const float* b_ih     = (const float*)d_in[4];
  const float* W_hh     = (const float*)d_in[5];
  const float* b_hh     = (const float*)d_in[6];
  const float* dec_bias = (const float*)d_in[7];
  const float* b_w      = (const float*)d_in[8];
  const float* b_h      = (const float*)d_in[9];
  const float* h0       = (const float*)d_in[10];
  float* out = (float*)d_out;

  auto alignup = [](size_t x) { return (x + 255) & ~(size_t)255; };
  char* p = (char*)d_ws;
  __bf16* emb    = (__bf16*)p; p += alignup((size_t)TTOT * KPAD * 2);
  __bf16* Wihp   = (__bf16*)p; p += alignup((size_t)NPAD * KPAD * 2);
  __bf16* Whhp   = (__bf16*)p; p += alignup((size_t)NPAD * NPAD * 2);
  float*  h_hist = (float*)p;  p += alignup((size_t)(SEQ + 1) * BSZ * NPAD * 4);
  float*  hU     = (float*)p;  p += alignup((size_t)TTOT * NPAD * 4);
  float*  scores = (float*)p;  p += alignup((size_t)TTOT * NCAND * 4);
  float*  h_flat = h_hist + (size_t)BSZ * NPAD;  // rows for steps 1..SEQ

  // Pack / gather / init
  {
    int tot = NPAD * KPAD;
    k_pack_w<<<(tot + 255) / 256, 256, 0, stream>>>(W_ih, Wihp, NHID, NINP, NPAD, KPAD);
  }
  {
    int tot = NPAD * NPAD;
    k_pack_w<<<(tot + 255) / 256, 256, 0, stream>>>(W_hh, Whhp, NHID, NHID, NPAD, NPAD);
  }
  {
    int tot = TTOT * KPAD;
    k_embed_pack<<<(tot + 255) / 256, 256, 0, stream>>>(data, enc, emb);
  }
  {
    int tot = BSZ * NPAD;
    k_h0_init<<<(tot + 255) / 256, 256, 0, stream>>>(h0, h_hist);
  }

  // Sequential ELU-RNN scan
  for (int s = 0; s < SEQ; ++s)
    k_rnn_step<<<NT, 64, 0, stream>>>(emb, Wihp, Whhp, b_ih, b_hh, h_hist, s);

  // hU = h_flat @ W_hh^T + b_hh
  k_hu_gemm<<<dim3(TTOT / 16, NT / 4), 128, 0, stream>>>(h_flat, Whhp, b_hh, hU);

  // Fused candidate GEMM + elu + distance + threshold/temperature/bias
  k_cand_score<<<MT_C / 2, 64, 0, stream>>>(cand, enc, Wihp, b_ih, hU, h_flat,
                                            b_w, b_h, dec_bias, scores);

  // log-softmax over candidates
  k_logsoftmax<<<TTOT / 4, 128, 0, stream>>>(scores, out);
}
